// InterestEvolutionLayer_52553219834437
// MI455X (gfx1250) — matrix operand order repaired and checked by reading, not compiled, not run on hardware
//
#include <hip/hip_runtime.h>
#include <math.h>

// Problem sizes (fixed by the reference)
#define B_   1024
#define S_   200
#define H_   256
#define A_   128
#define K2_  512   // I + H

typedef __bf16 v16bf __attribute__((ext_vector_type(16)));
typedef __bf16 v8bf  __attribute__((ext_vector_type(8)));
typedef float  v8f   __attribute__((ext_vector_type(8)));

// ---------------------------------------------------------------------------
// WMMA helpers (CDNA5 wave32, V_WMMA_F32_16X16X32_BF16)
//
// A-matrix (16x32 bf16): lane L<16 holds row L, K = {K0..K0+7, K0+16..K0+23};
//                        lane L>=16 holds row L-16, K = {K0+8..15, K0+24..31}.
// B-matrix (32x16 bf16): lane L holds column L%16, same K order as A:
//                        {K0..7, K0+16..23} (lanes 0-15), {K0+8..15, K0+24..31} (16-31).
// C/D (16x16 f32): c[v] = D[M = v + 8*(lane/16)][N = lane%16].
// ---------------------------------------------------------------------------
__device__ __forceinline__ v16bf load_a16(const __bf16* row, int K0, int lane) {
  const int koff = (lane < 16) ? 0 : 8;
  const __bf16* p = row + K0 + koff;
  v8bf lo = *(const v8bf*)(p);
  v8bf hi = *(const v8bf*)(p + 16);
  v16bf a;
#pragma unroll
  for (int i = 0; i < 8; ++i) { a[i] = lo[i]; a[i + 8] = hi[i]; }
  return a;
}

__device__ __forceinline__ v16bf load_b16(const __bf16* wrow, int K0, int lane) {
  const int kb = K0 + ((lane < 16) ? 0 : 8);
  v8bf lo = *(const v8bf*)(wrow + kb);
  v8bf hi = *(const v8bf*)(wrow + kb + 16);
  v16bf b;
#pragma unroll
  for (int i = 0; i < 8; ++i) { b[i] = lo[i]; b[i + 8] = hi[i]; }
  return b;
}

__device__ __forceinline__ v8f wmma_bf16(v16bf a, v16bf b, v8f c) {
  return __builtin_amdgcn_wmma_f32_16x16x32_bf16(false, a, false, b,
                                                 (short)0, c, false, false);
}

__device__ __forceinline__ float sigmoidf_dev(float x) {
  return 1.0f / (1.0f + __expf(-x));
}

// ---------------------------------------------------------------------------
// fp32 -> bf16 weight conversion (runs every call; idempotent/deterministic)
// ---------------------------------------------------------------------------
__global__ void convert_bf16_kernel(const float* __restrict__ src,
                                    __bf16* __restrict__ dst, int n) {
  int i = blockIdx.x * blockDim.x + threadIdx.x;
  if (i < n) dst[i] = (__bf16)src[i];
}

// ---------------------------------------------------------------------------
// Attention scores: score[b,s] = relu([hid|tgt] @ W1^T + b1) @ W2^T
// Block: 128 threads (4 waves), 32 rows of (b,s). K = 512, N = A = 128.
// Wave w: rows (w/2)*16 .. +15, columns (w%2)*64 .. +63 (4 N-subtiles).
// B-tiles batched per k-step so all loads are in flight before the WMMAs.
// ---------------------------------------------------------------------------
__global__ void attn_score_kernel(const float* __restrict__ hidden,
                                  const float* __restrict__ target,
                                  const __bf16* __restrict__ W1b,
                                  const float* __restrict__ b1,
                                  const float* __restrict__ W2,
                                  float* __restrict__ scores) {
  __shared__ __bf16 ta[32][520];   // 32 rows x 512 (pad 8: conflict-free)
  __shared__ float  ssc[32];

  const int tid    = threadIdx.x;
  const int rowblk = blockIdx.x * 32;

  if (tid < 32) ssc[tid] = 0.0f;

  // Stage combined [hidden | target] rows, fp32 -> bf16, float4 granularity.
  for (int idx = tid; idx < 32 * 128; idx += 128) {
    const int r  = idx >> 7;
    const int k4 = (idx & 127) << 2;
    const int rg = rowblk + r;
    const int b  = rg / S_;
    const int ss = rg - b * S_;
    float4 v;
    if (k4 < 256)
      v = *(const float4*)(hidden + ((size_t)b * S_ + ss) * H_ + k4);
    else
      v = *(const float4*)(target + (size_t)b * H_ + (k4 - 256));
    __bf16* d = &ta[r][k4];
    d[0] = (__bf16)v.x; d[1] = (__bf16)v.y;
    d[2] = (__bf16)v.z; d[3] = (__bf16)v.w;
  }
  __syncthreads();

  const int wave  = tid >> 5;
  const int lane  = tid & 31;
  const int mbase = (wave >> 1) << 4;   // 0 or 16
  const int nbase = (wave & 1) << 6;    // 0 or 64
  const int arow  = mbase + (lane & 15);

  v8f c[4] = {};
#pragma unroll
  for (int kk = 0; kk < 16; ++kk) {
    const int K0 = kk * 32;
    v16bf bm[4];
#pragma unroll
    for (int sub = 0; sub < 4; ++sub) {
      const int n = nbase + (sub << 4) + (lane & 15);
      bm[sub] = load_b16(W1b + (size_t)n * K2_, K0, lane);
    }
    v16bf a = load_a16(&ta[arow][0], K0, lane);
#pragma unroll
    for (int sub = 0; sub < 4; ++sub) c[sub] = wmma_bf16(a, bm[sub], c[sub]);
  }

  // ReLU(+b1) then dot with W2, reduce over the N dimension.
  float part[8];
#pragma unroll
  for (int v = 0; v < 8; ++v) part[v] = 0.0f;
#pragma unroll
  for (int sub = 0; sub < 4; ++sub) {
    const int n = nbase + (sub << 4) + (lane & 15);
    const float b1n = b1[n];
    const float w2n = W2[n];
#pragma unroll
    for (int v = 0; v < 8; ++v) {
      float t = c[sub][v] + b1n;
      t = t > 0.0f ? t : 0.0f;
      part[v] += t * w2n;
    }
  }
#pragma unroll
  for (int v = 0; v < 8; ++v) {
#pragma unroll
    for (int m = 1; m < 16; m <<= 1)
      part[v] += __shfl_xor(part[v], m, 16);
  }
  if ((lane & 15) == 0) {
    const int vrow = mbase + ((lane >> 4) << 3);
#pragma unroll
    for (int v = 0; v < 8; ++v) atomicAdd(&ssc[vrow + v], part[v]);
  }
  __syncthreads();
  if (tid < 32) scores[(size_t)rowblk + tid] = ssc[tid];
}

// ---------------------------------------------------------------------------
// Masked softmax over S per batch row (in place on scores buffer).
// ---------------------------------------------------------------------------
__global__ void softmax_kernel(float* __restrict__ att,
                               const int* __restrict__ lengths) {
  __shared__ float red[256];
  const int b = blockIdx.x, tid = threadIdx.x;
  const int len = lengths[b];
  float sc = -1e9f;
  if (tid < S_) {
    sc = att[(size_t)b * S_ + tid];
    if (tid >= len) sc = -1e9f;
  }
  red[tid] = (tid < S_) ? sc : -3.0e38f;
  __syncthreads();
  for (int o = 128; o > 0; o >>= 1) {
    if (tid < o) red[tid] = fmaxf(red[tid], red[tid + o]);
    __syncthreads();
  }
  const float mx = red[0];
  __syncthreads();
  float e = 0.0f;
  if (tid < S_) e = __expf(sc - mx);
  red[tid] = e;
  __syncthreads();
  for (int o = 128; o > 0; o >>= 1) {
    if (tid < o) red[tid] += red[tid + o];
    __syncthreads();
  }
  const float inv = 1.0f / red[0];
  if (tid < S_) att[(size_t)b * S_ + tid] = e * inv;
}

// ---------------------------------------------------------------------------
// Persistent AUGRU scan. Batch rows are fully independent in the recurrence,
// so each block owns 32 rows and runs ALL 200 steps locally: ONE kernel launch
// for the whole scan, h-state resident in LDS (fp32 + bf16 mirror), r*h and z
// tiles in LDS, 3 barriers per step. Global traffic per step: x-tile + weights.
//
// 32 blocks x 256 threads (8 waves).
// Phase 1 (N=512): wave w -> rows (w/4)*16, cols (w&3)*128 (8 subtiles).
//   r-waves write th = bf16(r*h); z-waves write zf = sigmoid(.)*att.
// Phase 2 (N=256): wave w -> rows (w/4)*16, cols (w&3)*64 (4 subtiles).
//   n = tanh(.); h = (1-z)*h + z*n, updated in LDS (fp32 + bf16).
// ---------------------------------------------------------------------------
__global__ void gru_scan_kernel(const float* __restrict__ hidden,
                                const float* __restrict__ att,
                                const __bf16* __restrict__ Wrb,
                                const __bf16* __restrict__ Wzb,
                                const __bf16* __restrict__ Wnb,
                                const float* __restrict__ br,
                                const float* __restrict__ bz,
                                const float* __restrict__ bn,
                                float* __restrict__ h_out) {
  __shared__ __bf16 tx[32][264];  // x tile (bf16, pad 8)
  __shared__ __bf16 th[32][264];  // (r*h) tile (bf16)
  __shared__ __bf16 hb[32][264];  // h state, bf16 mirror (WMMA A operand)
  __shared__ float  hf[32][264];  // h state, fp32 master
  __shared__ float  zf[32][264];  // z gate, fp32

  const int tid    = threadIdx.x;
  const int rowblk = blockIdx.x * 32;
  const int wave   = tid >> 5;
  const int lane   = tid & 31;

  const int mbase = (wave >> 2) << 4;    // 0 or 16 (both phases)
  const int arow  = mbase + (lane & 15);
  const int vrow  = mbase + ((lane >> 4) << 3);

  // Phase-1 wave role: cols (wave&3)*128 -> r (0..255) or z (256..511).
  const int  nbase1 = (wave & 3) << 7;
  const bool isZ    = nbase1 >= 256;
  const __bf16* Wg   = isZ ? Wzb : Wrb;
  const float*  bias = isZ ? bz : br;
  const int     nb   = isZ ? (nbase1 - 256) : nbase1;
  // Phase-2 cols.
  const int nbase2 = (wave & 3) << 6;

  // Init h state to zero.
  for (int idx = tid; idx < 32 * 256; idx += 256) {
    const int r = idx >> 8, k = idx & 255;
    hf[r][k] = 0.0f;
    hb[r][k] = (__bf16)0.0f;
  }

#pragma unroll 1
  for (int s = 0; s < S_; ++s) {
    __syncthreads();  // prev step's tx/th readers done; h state visible

    // Stage x = hidden[:, s, :] (fp32 -> bf16), float4 granularity.
    for (int idx = tid; idx < 32 * 64; idx += 256) {
      const int r  = idx >> 6;
      const int k4 = (idx & 63) << 2;
      float4 v = *(const float4*)(hidden +
                  ((size_t)(rowblk + r) * S_ + s) * H_ + k4);
      __bf16* d = &tx[r][k4];
      d[0] = (__bf16)v.x; d[1] = (__bf16)v.y;
      d[2] = (__bf16)v.z; d[3] = (__bf16)v.w;
    }
    __syncthreads();

    // ------------------ Phase 1: r and z gates (N = 512) ------------------
    {
      v8f c[8] = {};
#pragma unroll
      for (int kk = 0; kk < 8; ++kk) {        // x half (K 0..255)
        const int K0 = kk * 32;
        v16bf bm[8];
#pragma unroll
        for (int sub = 0; sub < 8; ++sub)
          bm[sub] = load_b16(Wg + (size_t)(nb + (sub << 4) + (lane & 15)) * K2_,
                             K0, lane);
        v16bf a = load_a16(&tx[arow][0], K0, lane);
#pragma unroll
        for (int sub = 0; sub < 8; ++sub) c[sub] = wmma_bf16(a, bm[sub], c[sub]);
      }
#pragma unroll
      for (int kk = 0; kk < 8; ++kk) {        // h half (K 256..511), LDS
        const int K0 = kk * 32;
        v16bf bm[8];
#pragma unroll
        for (int sub = 0; sub < 8; ++sub)
          bm[sub] = load_b16(Wg + (size_t)(nb + (sub << 4) + (lane & 15)) * K2_,
                             256 + K0, lane);
        v16bf a = load_a16(&hb[arow][0], K0, lane);
#pragma unroll
        for (int sub = 0; sub < 8; ++sub) c[sub] = wmma_bf16(a, bm[sub], c[sub]);
      }

#pragma unroll
      for (int v = 0; v < 8; ++v) {
        const int lr = vrow + v;
        const float aw =
            isZ ? att[(size_t)(rowblk + lr) * S_ + s] : 0.0f;
#pragma unroll
        for (int sub = 0; sub < 8; ++sub) {
          const int nl = nb + (sub << 4) + (lane & 15);
          const float sg = sigmoidf_dev(c[sub][v] + bias[nl]);
          if (isZ) zf[lr][nl] = sg * aw;
          else     th[lr][nl] = (__bf16)(sg * hf[lr][nl]);
        }
      }
    }
    __syncthreads();

    // ------------- Phase 2: candidate n + state update (N = 256) ----------
    {
      v8f c[4] = {};
#pragma unroll
      for (int kk = 0; kk < 8; ++kk) {        // x half
        const int K0 = kk * 32;
        v16bf bm[4];
#pragma unroll
        for (int sub = 0; sub < 4; ++sub)
          bm[sub] = load_b16(
              Wnb + (size_t)(nbase2 + (sub << 4) + (lane & 15)) * K2_, K0, lane);
        v16bf a = load_a16(&tx[arow][0], K0, lane);
#pragma unroll
        for (int sub = 0; sub < 4; ++sub) c[sub] = wmma_bf16(a, bm[sub], c[sub]);
      }
#pragma unroll
      for (int kk = 0; kk < 8; ++kk) {        // (r*h) half
        const int K0 = kk * 32;
        v16bf bm[4];
#pragma unroll
        for (int sub = 0; sub < 4; ++sub)
          bm[sub] = load_b16(
              Wnb + (size_t)(nbase2 + (sub << 4) + (lane & 15)) * K2_,
              256 + K0, lane);
        v16bf a = load_a16(&th[arow][0], K0, lane);
#pragma unroll
        for (int sub = 0; sub < 4; ++sub) c[sub] = wmma_bf16(a, bm[sub], c[sub]);
      }

#pragma unroll
      for (int v = 0; v < 8; ++v) {
        const int lr = vrow + v;
#pragma unroll
        for (int sub = 0; sub < 4; ++sub) {
          const int nl = nbase2 + (sub << 4) + (lane & 15);
          const float nv = tanhf(c[sub][v] + bn[nl]);
          const float z  = zf[lr][nl];
          const float hn = (1.0f - z) * hf[lr][nl] + z * nv;
          hf[lr][nl] = hn;
          hb[lr][nl] = (__bf16)hn;
        }
      }
    }
  }

  // Write final h to global output.
  __syncthreads();
  for (int idx = tid; idx < 32 * 256; idx += 256) {
    const int r = idx >> 8, k = idx & 255;
    h_out[(size_t)(rowblk + r) * H_ + k] = hf[r][k];
  }
}

// ---------------------------------------------------------------------------
// Host-side launcher
// ---------------------------------------------------------------------------
extern "C" void kernel_launch(void* const* d_in, const int* in_sizes, int n_in,
                              void* d_out, int out_size, void* d_ws,
                              size_t ws_size, hipStream_t stream) {
  const float* hidden = (const float*)d_in[0];
  const float* target = (const float*)d_in[1];
  const int*   lens   = (const int*)d_in[2];
  const float* W1 = (const float*)d_in[3];
  const float* b1 = (const float*)d_in[4];
  const float* W2 = (const float*)d_in[5];
  const float* Wr = (const float*)d_in[6];
  const float* br = (const float*)d_in[7];
  const float* Wz = (const float*)d_in[8];
  const float* bz = (const float*)d_in[9];
  const float* Wn = (const float*)d_in[10];
  const float* bn = (const float*)d_in[11];

  char* ws = (char*)d_ws;
  // Workspace layout (all offsets 256B-aligned):
  __bf16* W1b = (__bf16*)(ws + 0);        // 128*512 bf16 = 131072 B
  __bf16* Wrb = (__bf16*)(ws + 131072);   // 256*512 bf16 = 262144 B
  __bf16* Wzb = (__bf16*)(ws + 393216);   // 262144 B
  __bf16* Wnb = (__bf16*)(ws + 655360);   // 262144 B
  float*  att = (float*) (ws + 917504);   // B*S f32      = 819200 B
  float*  h   = (float*)d_out;            // final h (B*H f32)

  // 1) Convert weights fp32 -> bf16.
  convert_bf16_kernel<<<(A_ * K2_ + 255) / 256, 256, 0, stream>>>(W1, W1b,
                                                                  A_ * K2_);
  convert_bf16_kernel<<<(H_ * K2_ + 255) / 256, 256, 0, stream>>>(Wr, Wrb,
                                                                  H_ * K2_);
  convert_bf16_kernel<<<(H_ * K2_ + 255) / 256, 256, 0, stream>>>(Wz, Wzb,
                                                                  H_ * K2_);
  convert_bf16_kernel<<<(H_ * K2_ + 255) / 256, 256, 0, stream>>>(Wn, Wnb,
                                                                  H_ * K2_);

  // 2) Attention scores, then masked softmax (in place in `att`).
  attn_score_kernel<<<(B_ * S_) / 32, 128, 0, stream>>>(hidden, target, W1b,
                                                        b1, W2, att);
  softmax_kernel<<<B_, 256, 0, stream>>>(att, lens);

  // 3) Entire attention-gated GRU scan in ONE persistent kernel:
  //    batch rows are independent, h-state lives in LDS for all 200 steps.
  gru_scan_kernel<<<B_ / 32, 256, 0, stream>>>(hidden, att, Wrb, Wzb, Wnb, br,
                                               bz, bn, h);
}
